// LSTMAddonsDecoder_81939386073792
// MI455X (gfx1250) — matrix operand order repaired and checked by reading, not compiled, not run on hardware
//
#include <hip/hip_runtime.h>

typedef __bf16 bf16_t;
typedef __attribute__((ext_vector_type(8)))  __bf16 v8bf;
typedef __attribute__((ext_vector_type(16))) __bf16 v16bf;
typedef __attribute__((ext_vector_type(8)))  float  v8f;

#define B_   128
#define T_   512
#define D_   512
#define U_   1024
#define G_   4096   /* 4*U */
#define O_   512
#define KC_  1536   /* D + U */
#define KCH_ 128    /* K chunk staged in LDS per stage */
#define NST_ (KC_ / KCH_)   /* 12 stages */

// sched_group_barrier masks
#define SG_WMMA    0x008
#define SG_DSREAD  0x100

__device__ __forceinline__ bf16_t f2bf(float f) {
  unsigned u = __builtin_bit_cast(unsigned, f);
  unsigned r = u + 0x7FFFu + ((u >> 16) & 1u);   // round-to-nearest-even
  unsigned short h = (unsigned short)(r >> 16);
  return __builtin_bit_cast(bf16_t, h);
}

__device__ __forceinline__ float sigmoidf_(float x) {
  return 1.0f / (1.0f + __expf(-x));
}
__device__ __forceinline__ float tanhf_(float x) {
  float e = __expf(2.0f * x);
  return 1.0f - 2.0f / (e + 1.0f);
}

__device__ __forceinline__ v8bf ldv8(const bf16_t* p) {
  return *reinterpret_cast<const v8bf*>(p);
}
__device__ __forceinline__ v16bf cat16(v8bf a, v8bf b) {
  return __builtin_shufflevector(a, b, 0,1,2,3,4,5,6,7,8,9,10,11,12,13,14,15);
}
__device__ __forceinline__ v16bf ldfrag(const bf16_t* p) {
  return cat16(ldv8(p), ldv8(p + 8));
}

// ---------------- prep kernels ----------------

// x[B,T,D] f32  ->  xb[T,B,D] bf16  (per-step slab contiguous)
__global__ void k_convert_x(const float* __restrict__ x, bf16_t* __restrict__ xb) {
  size_t tid = (size_t)blockIdx.x * blockDim.x + threadIdx.x;
  size_t b = tid / ((size_t)T_ * D_);
  size_t r = tid % ((size_t)T_ * D_);
  size_t t = r / D_;
  size_t d = r % D_;
  xb[(t * B_ + b) * D_ + d] = f2bf(x[tid]);
}

// Wc[n][k] bf16, n in [0,4096), k in [0,1536): k<512 -> kernel[k][n], else rec[k-512][n]
__global__ void k_pack_w(const float* __restrict__ kern, const float* __restrict__ rec,
                         bf16_t* __restrict__ wc) {
  size_t tid = (size_t)blockIdx.x * blockDim.x + threadIdx.x;
  size_t n = tid / KC_;
  size_t k = tid % KC_;
  float v = (k < D_) ? kern[k * G_ + n] : rec[(k - D_) * G_ + n];
  wc[tid] = f2bf(v);
}

// Wdt[o][u] = Wd[u][o]  (bf16, N-major for B-fragment loads)
__global__ void k_pack_wd(const float* __restrict__ Wd, bf16_t* __restrict__ wdt) {
  size_t tid = (size_t)blockIdx.x * blockDim.x + threadIdx.x;
  size_t o = tid / U_;
  size_t u = tid % U_;
  wdt[tid] = f2bf(Wd[u * O_ + o]);
}

// hs slab 0 <- h0 (bf16);  c <- c0
__global__ void k_init_state(const float* __restrict__ h0, const float* __restrict__ c0,
                             bf16_t* __restrict__ hs0, float* __restrict__ c) {
  size_t tid = (size_t)blockIdx.x * blockDim.x + threadIdx.x;
  hs0[tid] = f2bf(h0[tid]);
  c[tid]   = c0[tid];
}

// ---------------- recurrent step: fused GEMM + gates, LDS-staged weights ----------------
// Block = one U tile (16 u's, all 4 gates); its 8 waves cover the 8 M tiles.
// Weight tile (64 rows x 128 K, 16 KB) is double-buffered in LDS (stage s+1
// globally prefetched into registers during stage s's WMMAs). Within a stage,
// sched_group_barrier pipelines the per-wave B-fragment ds_loads one iteration
// ahead of the WMMAs: pattern DS8 | (DS8, WMMA4) x 4.
// grid = 64 blocks x 256 threads.
__global__ void __launch_bounds__(256)
k_lstm_step(const bf16_t* __restrict__ xt,    // [B][D] bf16 (x slab for this t)
            const bf16_t* __restrict__ Wc,    // [4096][1536] bf16, N-major
            const float*  __restrict__ bias,  // [4096] f32
            const bf16_t* __restrict__ hprev, // [B][U] bf16 (h_{t-1}, hs slab t)
            bf16_t*       __restrict__ hnxt,  // [B][U] bf16 (h_t, hs slab t+1)
            float*        __restrict__ c) {   // [B][U] f32 in/out
  __shared__ bf16_t lb[2][64 * KCH_];   // 2 x 16 KB

  const int tid  = threadIdx.x;
  const int lane = tid & 31;
  const int wv   = tid >> 5;        // wave in block = M tile (0..7)
  const int ut   = blockIdx.x;      // U tile (0..63)
  const int half = lane >> 4;
  const int lm   = lane & 15;
  const int m0   = wv * 16;
  const int u0   = ut * 16;

  // cooperative staging identity: row sr = gate*16 + col, K sub-chunk sc
  const int sr = tid >> 2;          // 0..63
  const int sc = (tid & 3) * 32;    // 0,32,64,96
  const bf16_t* srow = Wc + (size_t)((sr >> 4) * U_ + u0 + (sr & 15)) * KC_ + sc;
  bf16_t* const sdst0 = &lb[0][sr * KCH_ + sc];
  bf16_t* const sdst1 = &lb[1][sr * KCH_ + sc];

  const v8f vz = {0.f,0.f,0.f,0.f,0.f,0.f,0.f,0.f};
  v8f acc0 = vz, acc1 = vz, acc2 = vz, acc3 = vz;

  const bf16_t* arowx = xt    + (size_t)(m0 + lm) * D_;
  const bf16_t* arowh = hprev + (size_t)(m0 + lm) * U_;

  // prologue: stage 0 into buffer 0
  v8bf st0 = ldv8(srow), st1 = ldv8(srow + 8), st2 = ldv8(srow + 16), st3 = ldv8(srow + 24);
  *(v8bf*)(sdst0)      = st0;  *(v8bf*)(sdst0 + 8)  = st1;
  *(v8bf*)(sdst0 + 16) = st2;  *(v8bf*)(sdst0 + 24) = st3;
  __syncthreads();

  int p = 0;
  #pragma unroll 1
  for (int s = 0; s < NST_; ++s) {
    const int kc = s * KCH_;
    // prefetch next weight stage into registers (overlaps with WMMAs below)
    if (s < NST_ - 1) {
      const bf16_t* nsrow = srow + kc + KCH_;
      st0 = ldv8(nsrow); st1 = ldv8(nsrow + 8); st2 = ldv8(nsrow + 16); st3 = ldv8(nsrow + 24);
      const bf16_t* arow_n = (kc + KCH_ < D_) ? (arowx + kc + KCH_) : (arowh + (kc + KCH_ - D_));
      __builtin_prefetch(arow_n, 0, 0);        // global_prefetch_b8
    }
    const bf16_t* arow = (kc < D_) ? (arowx + kc) : (arowh + (kc - D_));
    const bf16_t* lbase = &lb[p][(size_t)lm * KCH_ + 16 * half];

    // register pipeline: preload B fragments for kk = 0
    v16bf cb0 = ldfrag(lbase);
    v16bf cb1 = ldfrag(lbase + 16 * KCH_);
    v16bf cb2 = ldfrag(lbase + 32 * KCH_);
    v16bf cb3 = ldfrag(lbase + 48 * KCH_);
    __builtin_amdgcn_sched_group_barrier(SG_DSREAD, 8, 0);   // preload group

    #pragma unroll
    for (int kk = 0; kk < KCH_; kk += 32) {
      v16bf nb0 = cb0, nb1 = cb1, nb2 = cb2, nb3 = cb3;
      if (kk + 32 < KCH_) {
        const bf16_t* ln = lbase + kk + 32;
        nb0 = ldfrag(ln);
        nb1 = ldfrag(ln + 16 * KCH_);
        nb2 = ldfrag(ln + 32 * KCH_);
        nb3 = ldfrag(ln + 48 * KCH_);
      }
      v16bf a = ldfrag(arow + kk + 16 * half);
      acc0 = __builtin_amdgcn_wmma_f32_16x16x32_bf16(false, a, false, cb0, (short)0, acc0, false, false);
      acc1 = __builtin_amdgcn_wmma_f32_16x16x32_bf16(false, a, false, cb1, (short)0, acc1, false, false);
      acc2 = __builtin_amdgcn_wmma_f32_16x16x32_bf16(false, a, false, cb2, (short)0, acc2, false, false);
      acc3 = __builtin_amdgcn_wmma_f32_16x16x32_bf16(false, a, false, cb3, (short)0, acc3, false, false);
      cb0 = nb0; cb1 = nb1; cb2 = nb2; cb3 = nb3;
      // next iteration's 8 ds_loads ahead of this iteration's 4 WMMAs
      __builtin_amdgcn_sched_group_barrier(SG_DSREAD, 8, 0);
      __builtin_amdgcn_sched_group_barrier(SG_WMMA,   4, 0);
    }

    if (s < NST_ - 1) {
      bf16_t* d = (p == 0) ? sdst1 : sdst0;          // other buffer
      *(v8bf*)(d)      = st0;  *(v8bf*)(d + 8)  = st1;
      *(v8bf*)(d + 16) = st2;  *(v8bf*)(d + 24) = st3;
      __syncthreads();
      p ^= 1;
    }
  }

  // Gates; accumulator element r -> (b = m0 + r + 8*half, u = u0 + lm)
  const int u = u0 + lm;
  const float bi = bias[u];
  const float bf = bias[U_ + u];
  const float bg = bias[2 * U_ + u];
  const float bo = bias[3 * U_ + u];
  #pragma unroll
  for (int r = 0; r < 8; ++r) {
    const int b = m0 + r + 8 * half;
    const size_t idx = (size_t)b * U_ + u;
    float zi = acc0[r] + bi;
    float zf = acc1[r] + bf;
    float zg = acc2[r] + bg;
    float zo = acc3[r] + bo;
    float ig = sigmoidf_(zi);
    float fg = sigmoidf_(zf);
    float og = sigmoidf_(zo);
    float cn = fg * c[idx] + ig * tanhf_(zg);
    c[idx] = cn;
    hnxt[idx] = f2bf(og * tanhf_(cn));
  }
}

// ---------------- final projection: out = hs @ Wd + bd, one big GEMM ----------------
// hs: [T*B][U] bf16 (slabs 1..T of history). M = 65536, N = 512, K = 1024.
// Wave computes 1 M-tile x 4 N-tiles (A-fragment reuse). 32768 waves = 4096 blocks.
__global__ void __launch_bounds__(256)
k_proj_all(const bf16_t* __restrict__ hs,   // [T*B][U]
           const bf16_t* __restrict__ Wdt,  // [O][U]
           const float*  __restrict__ bd,   // [O]
           float*        __restrict__ out) {// [B][T][O]
  const int lane = threadIdx.x & 31;
  const int wid  = blockIdx.x * 8 + (threadIdx.x >> 5);
  const int q    = wid & 7;          // N quad -> n0 = q*64
  const int mt   = wid >> 3;         // 0..4095
  const int half = lane >> 4;
  const int lm   = lane & 15;
  const int m0   = mt * 16;
  const int n0   = q * 64;

  const v8f vz = {0.f,0.f,0.f,0.f,0.f,0.f,0.f,0.f};
  v8f acc[4] = {vz, vz, vz, vz};

  const bf16_t* arow = hs + (size_t)(m0 + lm) * U_;
  const bf16_t* br0 = Wdt + (size_t)(n0 + lm) * U_;
  const bf16_t* br1 = br0 + (size_t)16 * U_;
  const bf16_t* br2 = br0 + (size_t)32 * U_;
  const bf16_t* br3 = br0 + (size_t)48 * U_;

  #pragma unroll 4
  for (int k0 = 0; k0 < U_; k0 += 32) {
    v16bf a = ldfrag(arow + k0 + 16 * half);
    const int kb = k0 + 16 * half;
    v16bf b0 = ldfrag(br0 + kb);
    acc[0] = __builtin_amdgcn_wmma_f32_16x16x32_bf16(false, a, false, b0, (short)0, acc[0], false, false);
    v16bf b1 = ldfrag(br1 + kb);
    acc[1] = __builtin_amdgcn_wmma_f32_16x16x32_bf16(false, a, false, b1, (short)0, acc[1], false, false);
    v16bf b2 = ldfrag(br2 + kb);
    acc[2] = __builtin_amdgcn_wmma_f32_16x16x32_bf16(false, a, false, b2, (short)0, acc[2], false, false);
    v16bf b3 = ldfrag(br3 + kb);
    acc[3] = __builtin_amdgcn_wmma_f32_16x16x32_bf16(false, a, false, b3, (short)0, acc[3], false, false);
  }

  #pragma unroll
  for (int j = 0; j < 4; ++j) {
    const int o = n0 + j * 16 + lm;
    const float bdo = bd[o];
    #pragma unroll
    for (int r = 0; r < 8; ++r) {
      const int m = m0 + r + 8 * half;     // m = t*128 + b
      const int t = m >> 7;
      const int b = m & 127;
      out[((size_t)b * T_ + t) * O_ + o] = acc[j][r] + bdo;
    }
  }
}

// ---------------- launcher ----------------
extern "C" void kernel_launch(void* const* d_in, const int* in_sizes, int n_in,
                              void* d_out, int out_size, void* d_ws, size_t ws_size,
                              hipStream_t stream) {
  (void)in_sizes; (void)n_in; (void)out_size; (void)ws_size;
  const float* x    = (const float*)d_in[0];
  const float* h0   = (const float*)d_in[1];
  const float* c0   = (const float*)d_in[2];
  const float* kern = (const float*)d_in[3];
  const float* rec  = (const float*)d_in[4];
  const float* bias = (const float*)d_in[5];
  const float* Wd   = (const float*)d_in[6];
  const float* bd   = (const float*)d_in[7];
  float* out = (float*)d_out;

  const size_t SB = (size_t)B_ * U_;           // h slab elements

  char* ws = (char*)d_ws;
  bf16_t* xb  = (bf16_t*)ws;  ws += (size_t)B_ * T_ * D_ * 2;   // 64 MB
  bf16_t* Wc  = (bf16_t*)ws;  ws += (size_t)G_ * KC_ * 2;       // 12 MB
  bf16_t* Wdt = (bf16_t*)ws;  ws += (size_t)O_ * U_ * 2;        // 1 MB
  bf16_t* hs  = (bf16_t*)ws;  ws += (size_t)(T_ + 1) * SB * 2;  // 134.5 MB
  float*  cst = (float*)ws;   ws += SB * 4;                     // 512 KB

  k_convert_x <<<(B_ * T_ * D_) / 256, 256, 0, stream>>>(x, xb);
  k_pack_w    <<<(G_ * KC_)     / 256, 256, 0, stream>>>(kern, rec, Wc);
  k_pack_wd   <<<(O_ * U_)      / 256, 256, 0, stream>>>(Wd, Wdt);
  k_init_state<<<(B_ * U_)      / 256, 256, 0, stream>>>(h0, c0, hs, cst);

  for (int t = 0; t < T_; ++t) {
    const bf16_t* hprev = hs + (size_t)t * SB;
    bf16_t*       hnxt  = hs + (size_t)(t + 1) * SB;
    k_lstm_step<<<64, 256, 0, stream>>>(xb + (size_t)t * B_ * D_, Wc, bias, hprev, hnxt, cst);
  }

  k_proj_all<<<4096, 256, 0, stream>>>(hs + SB, Wdt, bd, out);
}
